// EDESC_81913616269499
// MI455X (gfx1250) — matrix operand order repaired and checked by reading, not compiled, not run on hardware
//
#include <hip/hip_runtime.h>
#include <hip/hip_bf16.h>

// Problem constants (from reference)
#define NROWS   16384
#define KDIM    4096
#define NCLUST  16
#define DSUB    256
#define ETA     5.0f
#define TEMP    0.1f
#define EPS_N   1e-8f

typedef __bf16 v16bf __attribute__((ext_vector_type(16)));
typedef float  v8f   __attribute__((ext_vector_type(8)));

union Frag16 {            // one WMMA 16-bit operand fragment: 16 halves / lane
    v16bf v;
    uint4 u[2];
};

// ---------------------------------------------------------------------------
// Kernel 1: per-row L2 normalize z, convert to bf16, write A (row-major).
// One block (256 threads) per row; thread t owns columns [16t, 16t+16).
// ---------------------------------------------------------------------------
__global__ __launch_bounds__(256)
void rownorm_convert_kernel(const float* __restrict__ z,
                            __hip_bfloat16* __restrict__ A) {
    const int row = blockIdx.x;
    const int t   = threadIdx.x;
    const float4* zr = (const float4*)(z + (size_t)row * KDIM) + t * 4;

    float4 v0 = zr[0], v1 = zr[1], v2 = zr[2], v3 = zr[3];
    float ss = v0.x*v0.x + v0.y*v0.y + v0.z*v0.z + v0.w*v0.w
             + v1.x*v1.x + v1.y*v1.y + v1.z*v1.z + v1.w*v1.w
             + v2.x*v2.x + v2.y*v2.y + v2.z*v2.z + v2.w*v2.w
             + v3.x*v3.x + v3.y*v3.y + v3.z*v3.z + v3.w*v3.w;

    __shared__ float red[256];
    red[t] = ss;
    __syncthreads();
    #pragma unroll
    for (int s = 128; s > 0; s >>= 1) {
        if (t < s) red[t] += red[t + s];
        __syncthreads();
    }
    const float inv = 1.0f / fmaxf(sqrtf(red[0]), EPS_N);

    float vs[16] = { v0.x, v0.y, v0.z, v0.w, v1.x, v1.y, v1.z, v1.w,
                     v2.x, v2.y, v2.z, v2.w, v3.x, v3.y, v3.z, v3.w };
    union { __hip_bfloat16 h[16]; uint4 u[2]; } pk;
    #pragma unroll
    for (int i = 0; i < 16; ++i) pk.h[i] = __float2bfloat16(vs[i] * inv);

    uint4* dst = (uint4*)(A + (size_t)row * KDIM + t * 16);
    dst[0] = pk.u[0];
    dst[1] = pk.u[1];
}

// ---------------------------------------------------------------------------
// Kernel 2: per-column inverse L2 norm of D.  4096 threads, coalesced rows.
// ---------------------------------------------------------------------------
__global__ __launch_bounds__(256)
void colnorm_kernel(const float* __restrict__ D, float* __restrict__ cinv) {
    const int j = blockIdx.x * 256 + threadIdx.x;
    float ss = 0.0f;
    #pragma unroll 4
    for (int k = 0; k < KDIM; ++k) {
        float v = D[(size_t)k * KDIM + j];
        ss += v * v;
    }
    cinv[j] = 1.0f / fmaxf(sqrtf(ss), EPS_N);
}

// ---------------------------------------------------------------------------
// Kernel 3: Bt[j][k] = bf16(D[k][j] * cinv[j])   (LDS-tiled 32x32 transpose)
// ---------------------------------------------------------------------------
__global__ __launch_bounds__(256)
void transpose_convert_kernel(const float* __restrict__ D,
                              const float* __restrict__ cinv,
                              __hip_bfloat16* __restrict__ Bt) {
    __shared__ float tile[32][33];
    const int j0 = blockIdx.x * 32;   // D column base
    const int k0 = blockIdx.y * 32;   // D row base
    const int tx = threadIdx.x, ty = threadIdx.y;

    #pragma unroll
    for (int i = 0; i < 4; ++i)
        tile[ty + i * 8][tx] = D[(size_t)(k0 + ty + i * 8) * KDIM + (j0 + tx)];
    __syncthreads();
    #pragma unroll
    for (int i = 0; i < 4; ++i) {
        const int j = j0 + ty + i * 8;
        Bt[(size_t)j * KDIM + (k0 + tx)] =
            __float2bfloat16(tile[tx][ty + i * 8] * cinv[j]);
    }
}

// ---------------------------------------------------------------------------
// Kernel 4: WMMA GEMM + fused square-sum per cluster.
// Grid: (NROWS/128, NCLUST). Block: 256 threads = 8 waves.
// Wave w owns rows [bx*128 + 16w, +16); block owns one cluster (256 cols).
// Double-buffered LDS for the B chunk (global loads overlap 16 WMMAs) and
// ping-ponged B-fragment registers (ds_load of tile nt+1 overlaps wmma nt).
// Accumulators: 16 tiles x v8f = 128 VGPRs per lane (wave32).
// ---------------------------------------------------------------------------
#define LB_STRIDE 40   // halves; 80B rows, 16B aligned
#define NKSTEPS   (KDIM / 32)

__global__ __launch_bounds__(256)
void gemm_cluster_kernel(const __hip_bfloat16* __restrict__ A,
                         const __hip_bfloat16* __restrict__ Bt,
                         float* __restrict__ s_out) {
    __shared__ __hip_bfloat16 lb[2][256 * LB_STRIDE];   // 2 x 20 KB

    const int t       = threadIdx.x;
    const int lane    = t & 31;
    const int wave    = t >> 5;
    const int cluster = blockIdx.y;
    const int n0      = cluster * DSUB;
    const int mtile   = blockIdx.x * 128 + wave * 16;
    const int lanelo  = lane & 15;
    const int hi      = lane >> 4;

    const __hip_bfloat16* Arow = A + (size_t)(mtile + lanelo) * KDIM;
    const __hip_bfloat16* Brow = Bt + (size_t)(n0 + t) * KDIM;

    v8f acc[16];
    #pragma unroll
    for (int nt = 0; nt < 16; ++nt) acc[nt] = (v8f)(0.0f);

    // Prologue: stage chunk 0 (thread t -> row n0+t, 64 bytes) into buffer 0.
    uint4 r0, r1, r2, r3;
    {
        const uint4* gs = (const uint4*)(Brow);
        r0 = gs[0]; r1 = gs[1]; r2 = gs[2]; r3 = gs[3];
        uint4* ld = (uint4*)(&lb[0][t * LB_STRIDE]);
        ld[0] = r0; ld[1] = r1; ld[2] = r2; ld[3] = r3;
    }

    for (int kt = 0; kt < NKSTEPS; ++kt) {
        const int cur = kt & 1;
        const int k0  = kt * 32;

        // Issue next chunk's global loads before the barrier; latency hides
        // under the 16 WMMAs below.
        if (kt + 1 < NKSTEPS) {
            const uint4* gs = (const uint4*)(Brow + k0 + 32);
            r0 = gs[0]; r1 = gs[1]; r2 = gs[2]; r3 = gs[3];
        }

        __syncthreads();   // lb[cur] fully stored; lb[cur^1] reads done

        // A fragment, 16-bit 16x32 layout:
        //   lanes 0-15:  K = k0+[0..15]
        //   lanes 16-31: K = k0+16+[0..15]
        Frag16 a;
        a.u[0] = *(const uint4*)(Arow + k0 + hi * 16);
        a.u[1] = *(const uint4*)(Arow + k0 + hi * 16 + 8);

        // 16 N-tiles with ping-ponged B-fragment registers.
        const __hip_bfloat16* lbc = &lb[cur][0];
        Frag16 b[2];
        {
            const uint4* ls =
                (const uint4*)(lbc + (size_t)lanelo * LB_STRIDE + hi * 16);
            b[0].u[0] = ls[0];
            b[0].u[1] = ls[1];
        }
        #pragma unroll
        for (int nt = 0; nt < 16; ++nt) {
            if (nt + 1 < 16) {
                const uint4* ls = (const uint4*)(
                    lbc + (size_t)((nt + 1) * 16 + lanelo) * LB_STRIDE + hi * 16);
                b[(nt + 1) & 1].u[0] = ls[0];
                b[(nt + 1) & 1].u[1] = ls[1];
            }
            acc[nt] = __builtin_amdgcn_wmma_f32_16x16x32_bf16(
                false, a.v, false, b[nt & 1].v, (short)0, acc[nt], false, false);
        }

        // Store the prefetched chunk into the other buffer (safe: all waves
        // finished reading it before this iteration's barrier).
        if (kt + 1 < NKSTEPS) {
            uint4* ld = (uint4*)(&lb[cur ^ 1][t * LB_STRIDE]);
            ld[0] = r0; ld[1] = r1; ld[2] = r2; ld[3] = r3;
        }
    }

    // Epilogue: s[m][cluster] = sum over 256 cols of sim^2.
    // C layout: VGPR r, lanes 0-15 -> M=r; lanes 16-31 -> M=r+8.
    float p[8];
    #pragma unroll
    for (int r = 0; r < 8; ++r) p[r] = 0.0f;
    #pragma unroll
    for (int nt = 0; nt < 16; ++nt) {
        #pragma unroll
        for (int r = 0; r < 8; ++r) {
            float x = acc[nt][r];
            p[r] += x * x;
        }
    }
    // reduce across the 16 lanes of each half-wave
    #pragma unroll
    for (int mask = 1; mask <= 8; mask <<= 1) {
        #pragma unroll
        for (int r = 0; r < 8; ++r) p[r] += __shfl_xor(p[r], mask, 32);
    }
    if (lanelo == 0) {
        const int mbase = mtile + hi * 8;
        #pragma unroll
        for (int r = 0; r < 8; ++r)
            s_out[(size_t)(mbase + r) * NCLUST + cluster] = p[r];
    }
}

// ---------------------------------------------------------------------------
// Kernel 5: copy z (f32) into output tail (overwrites the bf16-A staging).
// ---------------------------------------------------------------------------
__global__ __launch_bounds__(256)
void copy_z_kernel(const float* __restrict__ z, float* __restrict__ zout) {
    const size_t i = ((size_t)blockIdx.x * 256 + threadIdx.x) * 4;
    const float4 v = *(const float4*)(z + i);
    *(float4*)(zout + i) = v;
}

// ---------------------------------------------------------------------------
// Kernel 6: in-place softmax over 16 logits: v = ss/T + eta*dsub.
// ---------------------------------------------------------------------------
__global__ __launch_bounds__(256)
void softmax16_kernel(float* __restrict__ s) {
    const int row = blockIdx.x * 256 + threadIdx.x;
    float* p = s + (size_t)row * NCLUST;
    float v[16], m = -3.4e38f;
    #pragma unroll
    for (int i = 0; i < 16; ++i) {
        v[i] = p[i] * (1.0f / TEMP) + ETA * (float)DSUB;
        m = fmaxf(m, v[i]);
    }
    float sum = 0.0f;
    #pragma unroll
    for (int i = 0; i < 16; ++i) {
        v[i] = __expf(v[i] - m);
        sum += v[i];
    }
    const float inv = 1.0f / sum;
    #pragma unroll
    for (int i = 0; i < 16; ++i) p[i] = v[i] * inv;
}

// ---------------------------------------------------------------------------
extern "C" void kernel_launch(void* const* d_in, const int* in_sizes, int n_in,
                              void* d_out, int out_size, void* d_ws, size_t ws_size,
                              hipStream_t stream) {
    const float* z = (const float*)d_in[0];   // [16384, 4096] f32
    const float* D = (const float*)d_in[1];   // [4096, 4096]  f32

    float* s_out = (float*)d_out;                           // [16384, 16]
    float* z_out = (float*)d_out + (size_t)NROWS * NCLUST;  // [16384, 4096]

    // Stage bf16 A inside the (not-yet-written) z-copy output region (128 MB
    // of its 256 MB). It is overwritten by copy_z AFTER the GEMM completes.
    __hip_bfloat16* A = (__hip_bfloat16*)z_out;

    // Workspace: Bt (32 MB) + cinv (16 KB)
    char* ws = (char*)d_ws;
    __hip_bfloat16* Bt = (__hip_bfloat16*)ws;
    float* cinv = (float*)(ws + (size_t)KDIM * KDIM * sizeof(__hip_bfloat16));

    // 1) z row-normalize -> bf16 A
    rownorm_convert_kernel<<<NROWS, 256, 0, stream>>>(z, A);
    // 2) D column inverse norms
    colnorm_kernel<<<KDIM / 256, 256, 0, stream>>>(D, cinv);
    // 3) D -> Bt = (D * diag(cinv))^T in bf16
    transpose_convert_kernel<<<dim3(KDIM / 32, KDIM / 32), dim3(32, 8), 0, stream>>>(
        D, cinv, Bt);
    // 4) WMMA GEMM + fused per-cluster square-sum -> raw logits in s_out
    gemm_cluster_kernel<<<dim3(NROWS / 128, NCLUST), 256, 0, stream>>>(A, Bt, s_out);
    // 5) copy z into output tail (frees/overwrites bf16-A staging)
    copy_z_kernel<<<(NROWS * (size_t)KDIM / 4) / 256, 256, 0, stream>>>(z, z_out);
    // 6) softmax over 16 clusters, in place
    softmax16_kernel<<<NROWS / 256, 256, 0, stream>>>(s_out);
}
